// MultiHeadSelection_29652454211590
// MI455X (gfx1250) — matrix operand order repaired and checked by reading, not compiled, not run on hardware
//
#include <hip/hip_runtime.h>
#include <hip/hip_bf16.h>

// MI455X / gfx1250, wave32. WMMA bf16 16x16x32 everywhere.

typedef __attribute__((ext_vector_type(16))) __bf16 bf16x16;
typedef __attribute__((ext_vector_type(8)))  float  f32x8;

#define BB   16
#define LL   200
#define EE   300
#define HH   256
#define RR   50
#define RELD 64
#define MM   (BB*LL)   // 3200 rows (b*L + l)
#define MT   (MM/16)   // 200 row tiles

static __device__ __forceinline__ f32x8 wmma_bf16(bf16x16 a, bf16x16 b, f32x8 c) {
    return __builtin_amdgcn_wmma_f32_16x16x32_bf16(false, a, false, b, (short)0, c, false, false);
}

// gfx1250 async global->LDS copy (ASYNCcnt), per CDNA5 ISA.
// Inline asm used instead of the clang builtin for toolchain portability.
static __device__ __forceinline__ void async_copy_b128(unsigned lds_off, const void* gaddr) {
    asm volatile("global_load_async_to_lds_b128 %0, %1, off"
                 :: "v"(lds_off), "v"(gaddr)
                 : "memory");
}
static __device__ __forceinline__ void wait_async0() {
    asm volatile("s_wait_asynccnt 0x0" ::: "memory");
}

// ---------------------------------------------------------------------------
// Pack a weight matrix W (Nv rows x >=Kv cols, row stride ldw, col offset koff)
// into B-fragment layout:  Bpack[(kt*Ntiles + nt)*512 + lane*16 + e]
// where n = nt*16 + (lane&15), k = kt*32 + ((e>=8)?16:0) + (lane>>4)*8 + (e&7).  (ISA 7.12.2)
// ---------------------------------------------------------------------------
__global__ void k_packB(const float* __restrict__ W, __bf16* __restrict__ dst,
                        int ldw, int koff, int Ktiles, int Ntiles, int Kv, int Nv) {
    int idx = blockIdx.x * blockDim.x + threadIdx.x;
    int total = Ktiles * Ntiles * 512;
    if (idx >= total) return;
    int e    = idx & 15;
    int lane = (idx >> 4) & 31;
    int tile = idx >> 9;
    int nt = tile % Ntiles;
    int kt = tile / Ntiles;
    int n = nt * 16 + (lane & 15);
    int k = kt * 32 + ((e >= 8) ? 16 : 0) + (lane >> 4) * 8 + (e & 7);
    float v = 0.f;
    if (k < Kv && n < Nv) v = W[(size_t)n * ldw + koff + k];
    dst[idx] = (__bf16)v;
}

// ---------------------------------------------------------------------------
// Embedding gather -> A-fragment layout:
//   Apack[(mtile*Ktiles + kt)*512 + lane*16 + e]
//   m = mtile*16 + (lane&15), k = kt*32 + ((e>=8)?16:0) + (lane>>4)*8 + (e&7)
// ---------------------------------------------------------------------------
__global__ void k_embed(const int* __restrict__ tokens, const float* __restrict__ word_emb,
                        __bf16* __restrict__ xA) {
    const int Ktiles = 10; // K padded 300 -> 320
    int idx = blockIdx.x * blockDim.x + threadIdx.x;
    int total = MT * Ktiles * 512;
    if (idx >= total) return;
    int e    = idx & 15;
    int lane = (idx >> 4) & 31;
    int tile = idx >> 9;
    int kt = tile % Ktiles;
    int mtile = tile / Ktiles;
    int m = mtile * 16 + (lane & 15);
    int k = kt * 32 + ((e >= 8) ? 16 : 0) + (lane >> 4) * 8 + (e & 7);
    float v = 0.f;
    if (k < EE) {
        int tok = tokens[m];
        v = word_emb[(size_t)tok * EE + k];
    }
    xA[idx] = (__bf16)v;
}

__global__ void k_bias(const float* bihf, const float* bhhf,
                       const float* bihb, const float* bhhb,
                       float* biasF, float* biasB, float* accum) {
    int i = blockIdx.x * blockDim.x + threadIdx.x;
    if (i < 1024) { biasF[i] = bihf[i] + bhhf[i]; biasB[i] = bihb[i] + bhhb[i]; }
    if (i == 0) *accum = 0.f;
}

// ---------------------------------------------------------------------------
// Generic packed WMMA GEMM.  block=128 (4 waves), grid=(MT, ceil(Ntiles/4)).
// Output either f32 row-major (outF, row stride nstride) or re-packed as
// bf16 A-fragments for a following GEMM with K = Ntiles*16 (outA).
// ---------------------------------------------------------------------------
__global__ void k_gemm(const __bf16* __restrict__ Apack, const __bf16* __restrict__ Bpack,
                       const float* __restrict__ bias,
                       float* __restrict__ outF, int nstride,
                       __bf16* __restrict__ outA, int ktilesNext,
                       int Ktiles, int Ntiles, int doRelu) {
    int wv   = threadIdx.x >> 5;
    int lane = threadIdx.x & 31;
    int mtile = blockIdx.x;
    int ntile = blockIdx.y * 4 + wv;
    if (ntile >= Ntiles) return;
    f32x8 acc = {};
    const __bf16* aB = Apack + ((size_t)mtile * Ktiles) * 512 + lane * 16;
    for (int kt = 0; kt < Ktiles; ++kt) {
        bf16x16 a = *(const bf16x16*)(aB + (size_t)kt * 512);
        bf16x16 b = *(const bf16x16*)(Bpack + ((size_t)kt * Ntiles + ntile) * 512 + lane * 16);
        acc = wmma_bf16(a, b, acc);
    }
    int nl = lane & 15;
    int hf = lane >> 4;
    int n  = ntile * 16 + nl;
    float bv = bias ? bias[n] : 0.f;
#pragma unroll
    for (int v = 0; v < 8; ++v) {
        float x = acc[v] + bv;
        if (doRelu) x = fmaxf(x, 0.f);
        int ml = v + 8 * hf;
        if (outF) {
            outF[(size_t)(mtile * 16 + ml) * nstride + n] = x;
        } else {
            int kt2 = n >> 5, kp = n & 31;
            int e  = ((kp >> 4) << 3) + (kp & 7);
            int lt = (((kp >> 3) & 1) << 4) + ml;
            outA[(((size_t)mtile * ktilesNext + kt2) * 32 + lt) * 16 + e] = (__bf16)x;
        }
    }
}

// ---------------------------------------------------------------------------
// Persistent LSTM recurrence. grid=2 (fwd/bwd), block=512 (16 waves).
// Wave w owns h-columns [w*16, w*16+16): gate tiles {w, 16+w, 32+w, 48+w}.
// h recirculated via 8KB LDS A-fragment buffer; c lives in VGPRs.
// Per step: 8 ktiles x 4 gates = 32 v_wmma per wave.
// ---------------------------------------------------------------------------
__global__ void k_lstm(const float* __restrict__ preF, const float* __restrict__ preB,
                       const __bf16* __restrict__ WhhFp, const __bf16* __restrict__ WhhBp,
                       float* __restrict__ ofp, float* __restrict__ obp) {
    int dir = blockIdx.x;
    const float*  pre  = dir ? preB  : preF;
    const __bf16* Wp   = dir ? WhhBp : WhhFp;
    float*        outH = dir ? obp   : ofp;

    __shared__ __bf16 hA[8 * 512]; // 8 ktiles of packed A fragments (16x256 bf16)
    int tid  = threadIdx.x;
    int wv   = tid >> 5;
    int lane = tid & 31;
    int nl   = lane & 15;
    int hf   = lane >> 4;

    for (int i = tid; i < 8 * 512; i += 512) hA[i] = (__bf16)0.f;
    f32x8 cst = {};
    __syncthreads();

    for (int step = 0; step < LL; ++step) {
        int t = dir ? (LL - 1 - step) : step;
        int nI = wv * 16 + nl; // h-column within [0,256)
        f32x8 gi, gf, gg, go;
#pragma unroll
        for (int v = 0; v < 8; ++v) {
            int m = v + 8 * hf; // batch row
            size_t row = ((size_t)m * LL + t) * 1024;
            gi[v] = pre[row + 0   + nI];
            gf[v] = pre[row + 256 + nI];
            gg[v] = pre[row + 512 + nI];
            go[v] = pre[row + 768 + nI];
        }
        for (int kt = 0; kt < 8; ++kt) {
            bf16x16 a  = *(const bf16x16*)(hA + kt * 512 + lane * 16);
            bf16x16 b0 = *(const bf16x16*)(Wp + ((size_t)kt * 64 + ( 0 + wv)) * 512 + lane * 16);
            bf16x16 b1 = *(const bf16x16*)(Wp + ((size_t)kt * 64 + (16 + wv)) * 512 + lane * 16);
            bf16x16 b2 = *(const bf16x16*)(Wp + ((size_t)kt * 64 + (32 + wv)) * 512 + lane * 16);
            bf16x16 b3 = *(const bf16x16*)(Wp + ((size_t)kt * 64 + (48 + wv)) * 512 + lane * 16);
            gi = wmma_bf16(a, b0, gi);
            gf = wmma_bf16(a, b1, gf);
            gg = wmma_bf16(a, b2, gg);
            go = wmma_bf16(a, b3, go);
        }
        __syncthreads(); // all waves finished reading hA for this step
#pragma unroll
        for (int v = 0; v < 8; ++v) {
            float i_ = 1.f / (1.f + __expf(-gi[v]));
            float f_ = 1.f / (1.f + __expf(-gf[v]));
            float g_ = tanhf(gg[v]);
            float o_ = 1.f / (1.f + __expf(-go[v]));
            float c  = f_ * cst[v] + i_ * g_;
            cst[v] = c;
            float h = o_ * tanhf(c);
            int m = v + 8 * hf;       // batch
            int j = wv * 16 + nl;     // h-column
            outH[((size_t)m * LL + t) * HH + j] = h;
            int kt2 = j >> 5, kp = j & 31;
            int e  = ((kp >> 4) << 3) + (kp & 7);
            int lt = (((kp >> 3) & 1) << 4) + m;
            hA[(kt2 * 32 + lt) * 16 + e] = (__bf16)h;
        }
        __syncthreads(); // hA ready for next step
    }
}

// o = 0.5*(of+ob) -> f32 row-major + A-fragment pack (Ktiles=8 for K=256)
__global__ void k_post(const float* __restrict__ ofp, const float* __restrict__ obp,
                       float* __restrict__ o, __bf16* __restrict__ oA) {
    int idx = blockIdx.x * blockDim.x + threadIdx.x;
    if (idx >= MM * HH) return;
    int m = idx >> 8, j = idx & 255;
    float v = 0.5f * (ofp[idx] + obp[idx]);
    o[idx] = v;
    int mtile = m >> 4, ml = m & 15;
    int kt = j >> 5, kp = j & 31;
    int e  = ((kp >> 4) << 3) + (kp & 7);
    int lt = (((kp >> 3) & 1) << 4) + ml;
    oA[(((size_t)mtile * 8 + kt) * 32 + lt) * 16 + e] = (__bf16)v;
}

// emi = o @ W_emi^T + b_emi   (N=3, trivially VALU-bound)
__global__ void k_emi(const float* __restrict__ o, const float* __restrict__ W,
                      const float* __restrict__ bvec, float* __restrict__ emi) {
    int m = blockIdx.x * blockDim.x + threadIdx.x;
    if (m >= MM) return;
    float a0 = bvec[0], a1 = bvec[1], a2 = bvec[2];
    const float* row = o + (size_t)m * HH;
    for (int j = 0; j < HH; ++j) {
        float x = row[j];
        a0 += x * W[0 * HH + j];
        a1 += x * W[1 * HH + j];
        a2 += x * W[2 * HH + j];
    }
    emi[m * 3 + 0] = a0; emi[m * 3 + 1] = a1; emi[m * 3 + 2] = a2;
}

// ---------------------------------------------------------------------------
// Fused selection loss. One block per (b,i).  up[b,:,:] (200x64 f32 = 51.2KB)
// is staged into LDS with gfx1250 async global->LDS copies (ASYNCcnt), then
// uv tiles are built from LDS, WMMA'd against packed rel_emb (K=64), and the
// softplus/gold loss is reduced per block with one f32 atomic.
// uv / logits tensors (160MB) never touch HBM.
// ---------------------------------------------------------------------------
__global__ void k_sel(const float* __restrict__ up, const float* __restrict__ vpp,
                      const float* __restrict__ buv, const __bf16* __restrict__ relB,
                      const float* __restrict__ gold, float* __restrict__ accum) {
    int b = blockIdx.x / LL, i = blockIdx.x % LL;
    __shared__ __align__(16) float uS[LL * 64];   // 51.2 KB
    __shared__ __align__(16) float vrow[64];
    __shared__ float red[256];
    int tid = threadIdx.x;

    // async stage: 12800 floats = 3200 x b128 transfers across 256 threads
    {
        const float* src = up + (size_t)b * LL * 64;
        unsigned ldsBase = (unsigned)(unsigned long long)(void*)uS;
        for (int chunk = tid; chunk < (LL * 64) / 4; chunk += 256) {
            async_copy_b128(ldsBase + chunk * 16, src + chunk * 4);
        }
    }
    if (tid < 64) vrow[tid] = vpp[((size_t)(b * LL + i)) * 64 + tid] + buv[tid];
    wait_async0();
    __syncthreads();

    int wv = tid >> 5, lane = tid & 31, nl = lane & 15, hf = lane >> 4;
    float lsum = 0.f;
    const int JT = 13; // ceil(200/16)
    for (int task = wv; task < JT * 4; task += 8) {
        int jt = task >> 2, nt = task & 3;
        int j = jt * 16 + nl;       // A row (M dim = j)
        bool valid = j < LL;
        // prefetch the gold row this task will consume
        __builtin_prefetch(&gold[(((size_t)(b * LL + i)) * RR + nt * 16) * LL + jt * 16], 0, 0);
        f32x8 acc = {};
        for (int kk = 0; kk < 2; ++kk) {
            bf16x16 a;
            int kb = kk * 32 + hf * 8;
            if (valid) {
                const float* rp = uS + j * 64 + kb;   // LDS
                const float* vr = vrow + kb;          // LDS
#pragma unroll
                for (int q = 0; q < 8; ++q) {
                    a[q]     = (__bf16)fmaxf(rp[q]      + vr[q],      0.f);
                    a[q + 8] = (__bf16)fmaxf(rp[q + 16] + vr[q + 16], 0.f);
                }
            } else {
#pragma unroll
                for (int q = 0; q < 16; ++q) a[q] = (__bf16)0.f;
            }
            bf16x16 bf_ = *(const bf16x16*)(relB + ((size_t)(kk * 4 + nt) * 32 + lane) * 16);
            acc = wmma_bf16(a, bf_, acc);
        }
        int r = nt * 16 + nl;
#pragma unroll
        for (int v = 0; v < 8; ++v) {
            int jj = jt * 16 + v + 8 * hf;
            if (r < RR && jj < LL) {
                float x = acc[v];
                float g = gold[(((size_t)(b * LL + i)) * RR + r) * LL + jj];
                lsum += fmaxf(x, 0.f) + log1pf(__expf(-fabsf(x))) - x * g;
            }
        }
    }
    red[tid] = lsum;
    __syncthreads();
    for (int s = 128; s > 0; s >>= 1) {
        if (tid < s) red[tid] += red[tid + s];
        __syncthreads();
    }
    if (tid == 0) atomicAdd(accum, red[0]);
}

// CRF log-likelihood (T=3) + final combine. One block, 16 active lanes.
__global__ void k_crf(const float* __restrict__ emi, const int* __restrict__ bio,
                      const int* __restrict__ tokens,
                      const float* __restrict__ cs, const float* __restrict__ ce,
                      const float* __restrict__ ct,
                      const float* __restrict__ accum, float* __restrict__ out) {
    __shared__ float part[16];
    int b = threadIdx.x;
    if (b < 16) {
        float tr[3][3];
        for (int a = 0; a < 3; ++a)
            for (int c = 0; c < 3; ++c) tr[a][c] = ct[a * 3 + c];
        int tag0 = bio[b * LL];
        float alpha[3], e0[3];
        for (int s = 0; s < 3; ++s) {
            e0[s] = emi[((size_t)b * LL) * 3 + s];
            alpha[s] = cs[s] + e0[s];
        }
        float num = cs[tag0] + e0[tag0];
        int prevTag = tag0;
        int msum = (tokens[b * LL] != 0) ? 1 : 0;
        for (int t = 1; t < LL; ++t) {
            float et[3];
            for (int s = 0; s < 3; ++s) et[s] = emi[((size_t)b * LL + t) * 3 + s];
            int tg = bio[b * LL + t];
            float mt = (tokens[b * LL + t] != 0) ? 1.f : 0.f;
            msum += (int)mt;
            num += (tr[prevTag][tg] + et[tg]) * mt;
            prevTag = tg;
            float na[3];
            for (int s2 = 0; s2 < 3; ++s2) {
                float x0 = alpha[0] + tr[0][s2];
                float x1 = alpha[1] + tr[1][s2];
                float x2 = alpha[2] + tr[2][s2];
                float m = fmaxf(fmaxf(x0, x1), x2);
                na[s2] = m + __logf(__expf(x0 - m) + __expf(x1 - m) + __expf(x2 - m)) + et[s2];
            }
            if (mt > 0.5f) { alpha[0] = na[0]; alpha[1] = na[1]; alpha[2] = na[2]; }
        }
        int lastIdx = msum - 1;
        num += ce[bio[b * LL + lastIdx]];
        float a0 = alpha[0] + ce[0], a1 = alpha[1] + ce[1], a2 = alpha[2] + ce[2];
        float m = fmaxf(fmaxf(a0, a1), a2);
        float logZ = m + __logf(__expf(a0 - m) + __expf(a1 - m) + __expf(a2 - m));
        part[b] = num - logZ;
    }
    __syncthreads();
    if (threadIdx.x == 0) {
        float s = 0.f;
        for (int q = 0; q < 16; ++q) s += part[q];
        out[0] = s + accum[0] / 32000000.0f; // + mean(sel_loss)
    }
}

// ---------------------------------------------------------------------------
extern "C" void kernel_launch(void* const* d_in, const int* in_sizes, int n_in,
                              void* d_out, int out_size, void* d_ws, size_t ws_size,
                              hipStream_t stream) {
    const int*   tokens   = (const int*)  d_in[0];
    const int*   bio      = (const int*)  d_in[1];
    const float* gold     = (const float*)d_in[2];
    const float* word_emb = (const float*)d_in[3];
    const float* rel_emb  = (const float*)d_in[4];
    const float* Wih_f    = (const float*)d_in[5];
    const float* Whh_f    = (const float*)d_in[6];
    const float* bih_f    = (const float*)d_in[7];
    const float* bhh_f    = (const float*)d_in[8];
    const float* Wih_b    = (const float*)d_in[9];
    const float* Whh_b    = (const float*)d_in[10];
    const float* bih_b    = (const float*)d_in[11];
    const float* bhh_b    = (const float*)d_in[12];
    const float* W_emi    = (const float*)d_in[13];
    const float* b_emi    = (const float*)d_in[14];
    const float* Wu       = (const float*)d_in[15];
    const float* bu       = (const float*)d_in[16];
    const float* Wv       = (const float*)d_in[17];
    const float* bv       = (const float*)d_in[18];
    const float* Wuv      = (const float*)d_in[19];
    const float* buv      = (const float*)d_in[20];
    const float* crf_s    = (const float*)d_in[21];
    const float* crf_e    = (const float*)d_in[22];
    const float* crf_t    = (const float*)d_in[23];
    float* out = (float*)d_out;

    char* wp = (char*)d_ws;
    auto alloc = [&](size_t bytes) -> void* {
        void* p = (void*)wp;
        wp += (bytes + 255) & ~(size_t)255;
        return p;
    };
    __bf16* xA     = (__bf16*)alloc((size_t)MT * 10 * 512 * 2);
    __bf16* WihFp  = (__bf16*)alloc((size_t)10 * 64 * 512 * 2);
    __bf16* WihBp  = (__bf16*)alloc((size_t)10 * 64 * 512 * 2);
    __bf16* WhhFp  = (__bf16*)alloc((size_t)8 * 64 * 512 * 2);
    __bf16* WhhBp  = (__bf16*)alloc((size_t)8 * 64 * 512 * 2);
    float*  biasF  = (float*) alloc(1024 * 4);
    float*  biasB  = (float*) alloc(1024 * 4);
    float*  pre_f  = (float*) alloc((size_t)MM * 1024 * 4);
    float*  pre_b  = (float*) alloc((size_t)MM * 1024 * 4);
    float*  ofp    = (float*) alloc((size_t)MM * HH * 4);
    float*  obp    = (float*) alloc((size_t)MM * HH * 4);
    float*  o      = (float*) alloc((size_t)MM * HH * 4);
    __bf16* oA     = (__bf16*)alloc((size_t)MT * 8 * 512 * 2);
    __bf16* Wup    = (__bf16*)alloc((size_t)8 * 4 * 512 * 2);
    __bf16* Wvp    = (__bf16*)alloc((size_t)8 * 4 * 512 * 2);
    __bf16* uA     = (__bf16*)alloc((size_t)MT * 2 * 512 * 2);
    __bf16* vA     = (__bf16*)alloc((size_t)MT * 2 * 512 * 2);
    __bf16* WuvAp  = (__bf16*)alloc((size_t)2 * 4 * 512 * 2);
    __bf16* WuvBp  = (__bf16*)alloc((size_t)2 * 4 * 512 * 2);
    __bf16* relp   = (__bf16*)alloc((size_t)2 * 4 * 512 * 2);
    float*  up     = (float*) alloc((size_t)MM * 64 * 4);
    float*  vpbuf  = (float*) alloc((size_t)MM * 64 * 4);
    float*  emi    = (float*) alloc((size_t)MM * 3 * 4);
    float*  accum  = (float*) alloc(256);

    auto cdiv = [](int a, int b) { return (a + b - 1) / b; };

    // ---- weight packing + bias + accumulator init
    k_bias<<<4, 256, 0, stream>>>(bih_f, bhh_f, bih_b, bhh_b, biasF, biasB, accum);
    k_packB<<<cdiv(10 * 64 * 512, 256), 256, 0, stream>>>(Wih_f, WihFp, 300, 0, 10, 64, 300, 1024);
    k_packB<<<cdiv(10 * 64 * 512, 256), 256, 0, stream>>>(Wih_b, WihBp, 300, 0, 10, 64, 300, 1024);
    k_packB<<<cdiv(8 * 64 * 512, 256), 256, 0, stream>>>(Whh_f, WhhFp, 256, 0, 8, 64, 256, 1024);
    k_packB<<<cdiv(8 * 64 * 512, 256), 256, 0, stream>>>(Whh_b, WhhBp, 256, 0, 8, 64, 256, 1024);
    k_packB<<<cdiv(8 * 4 * 512, 256), 256, 0, stream>>>(Wu, Wup, 256, 0, 8, 4, 256, 64);
    k_packB<<<cdiv(8 * 4 * 512, 256), 256, 0, stream>>>(Wv, Wvp, 256, 0, 8, 4, 256, 64);
    k_packB<<<cdiv(2 * 4 * 512, 256), 256, 0, stream>>>(Wuv, WuvAp, 128, 0, 2, 4, 64, 64);
    k_packB<<<cdiv(2 * 4 * 512, 256), 256, 0, stream>>>(Wuv, WuvBp, 128, 64, 2, 4, 64, 64);
    k_packB<<<cdiv(2 * 4 * 512, 256), 256, 0, stream>>>(rel_emb, relp, 64, 0, 2, 4, 64, 50);

    // ---- embedding gather (A-pack)
    k_embed<<<cdiv(MT * 10 * 512, 256), 256, 0, stream>>>(tokens, word_emb, xA);

    // ---- pre = x @ Wih^T + (bih+bhh)  (M=3200, N=1024, K=320)
    k_gemm<<<dim3(MT, 16), 128, 0, stream>>>(xA, WihFp, biasF, pre_f, 1024, nullptr, 0, 10, 64, 0);
    k_gemm<<<dim3(MT, 16), 128, 0, stream>>>(xA, WihBp, biasB, pre_b, 1024, nullptr, 0, 10, 64, 0);

    // ---- persistent bidirectional LSTM recurrence
    k_lstm<<<2, 512, 0, stream>>>(pre_f, pre_b, WhhFp, WhhBp, ofp, obp);

    // ---- o = 0.5*(of+ob), pack for K=256 GEMMs
    k_post<<<cdiv(MM * HH, 256), 256, 0, stream>>>(ofp, obp, o, oA);

    // ---- u = relu(o@Wu^T + bu), v = relu(o@Wv^T + bv)  -> A-packs (K=64 next)
    k_gemm<<<dim3(MT, 1), 128, 0, stream>>>(oA, Wup, bu, nullptr, 0, uA, 2, 8, 4, 1);
    k_gemm<<<dim3(MT, 1), 128, 0, stream>>>(oA, Wvp, bv, nullptr, 0, vA, 2, 8, 4, 1);

    // ---- up = u@WuvA^T, vp = v@WuvB^T (f32, no bias/relu)
    k_gemm<<<dim3(MT, 1), 128, 0, stream>>>(uA, WuvAp, nullptr, up, 64, nullptr, 0, 2, 4, 0);
    k_gemm<<<dim3(MT, 1), 128, 0, stream>>>(vA, WuvBp, nullptr, vpbuf, 64, nullptr, 0, 2, 4, 0);

    // ---- fused selection logits + softplus loss (atomic accumulate)
    k_sel<<<BB * LL, 256, 0, stream>>>(up, vpbuf, buv, relp, gold, accum);

    // ---- emissions + CRF + final combine
    k_emi<<<cdiv(MM, 256), 256, 0, stream>>>(o, W_emi, b_emi, emi);
    k_crf<<<1, 32, 0, stream>>>(emi, bio, tokens, crf_s, crf_e, crf_t, accum, out);
}
